// MultiHeadAttention_41120016892169
// MI455X (gfx1250) — matrix operand order repaired and checked by reading, not compiled, not run on hardware
//
#include <hip/hip_runtime.h>
#include <hip/hip_bf16.h>

// ---------------------------------------------------------------------------
// MultiHeadAttention for MI455X (gfx1250): bf16 WMMA + TDM/async LDS staging.
// B=2, S=2048, D=1024, H=16, HD=64.
// ---------------------------------------------------------------------------

typedef __attribute__((ext_vector_type(16))) __bf16 v16bf;
typedef __attribute__((ext_vector_type(8)))  __bf16 v8bf;
typedef __attribute__((ext_vector_type(8)))  float  v8f;
typedef __attribute__((ext_vector_type(4)))  float  v4f;
typedef __attribute__((ext_vector_type(4)))  unsigned int u32x4;
typedef __attribute__((ext_vector_type(8)))  int    i32x8;
typedef __attribute__((ext_vector_type(4)))  int    i32x4;

#define BQ 2
#define SQ 2048
#define DQ 1024
#define HQ 16
#define HDQ 64

// ---- gfx1250 data-mover feature detection ---------------------------------
#if defined(__AMDGCN__) && __has_builtin(__builtin_amdgcn_tensor_load_to_lds) && \
    __has_builtin(__builtin_amdgcn_s_wait_tensorcnt)
#define USE_TDM 1
#else
#define USE_TDM 0
#endif
#if defined(__AMDGCN__) &&                                        \
    __has_builtin(__builtin_amdgcn_global_load_async_to_lds_b128) && \
    __has_builtin(__builtin_amdgcn_s_wait_asynccnt)
#define USE_ASYNC 1
#else
#define USE_ASYNC 0
#endif

static __device__ __forceinline__ unsigned lds_off(const void* p) {
  // generic VA of LDS: low 32 bits are the LDS byte address
  return (unsigned)(uintptr_t)p;
}

#if USE_TDM
// 2D tile load via Tensor Data Mover (6-arg builtin: g0,g1,g2,g3,extra,cpol).
// dims/strides in elements of 2 bytes (data_size code 1).
// padIntervalCode: pad after 2^(c+1) DWORDs; padAmountCode: pad (c+1) DWORDs.
static __device__ __forceinline__ void tdm_load_2d(
    const void* gsrc, unsigned lds_byte_off, unsigned dim0, unsigned dim1,
    unsigned stride0, unsigned tileW, unsigned tileH, unsigned padIntervalCode,
    unsigned padAmountCode) {
  unsigned long long ga = (unsigned long long)(uintptr_t)gsrc;
  u32x4 g0;
  g0[0] = 1u;                                      // count=1 (valid), user mode
  g0[1] = lds_byte_off;                            // lds_addr [63:32]
  g0[2] = (unsigned)ga;                            // global_addr [95:64]
  g0[3] = (unsigned)((ga >> 32) & 0x01ffffffu) | (2u << 30);  // [120:96]|type=2
  i32x8 g1;
  g1[0] = (int)((1u << 16) | (1u << 20) | (padIntervalCode << 22) |
                (padAmountCode << 25));            // data_size=2B, pad_enable
  g1[1] = (int)((dim0 & 0xffffu) << 16);           // tensor_dim0 lo -> [63:48]
  g1[2] = (int)(((dim0 >> 16) & 0xffffu) | ((dim1 & 0xffffu) << 16));
  g1[3] = (int)(((dim1 >> 16) & 0xffffu) | ((tileW & 0xffffu) << 16));
  g1[4] = (int)(tileH & 0xffffu);                  // tile_dim1 (tile_dim2=0)
  g1[5] = (int)stride0;                            // tensor_dim0_stride[31:0]
  g1[6] = 0;
  g1[7] = 0;
  i32x4 z4 = (i32x4)(0);
  i32x8 z8 = (i32x8)(0);
  __builtin_amdgcn_tensor_load_to_lds(g0, g1, z4, z4, z8, 0);
}
#endif

// per-lane 16B copy global->LDS (async if available)
static __device__ __forceinline__ void cp16(__bf16* lds_dst,
                                            const __bf16* gsrc) {
#if USE_ASYNC
  __builtin_amdgcn_global_load_async_to_lds_b128(
      (__attribute__((address_space(1))) i32x4*)(uintptr_t)gsrc,
      (__attribute__((address_space(3))) i32x4*)(unsigned)(uintptr_t)lds_dst,
      0, 0);
#else
  *(v8bf*)lds_dst = *(const v8bf*)gsrc;
#endif
}

static __device__ __forceinline__ void stage_wait() {
#if USE_TDM
  __builtin_amdgcn_s_wait_tensorcnt(0);
#elif USE_ASYNC
  __builtin_amdgcn_s_wait_asynccnt(0);
#endif
}

static __device__ __forceinline__ v16bf mk16(v8bf lo, v8bf hi) {
  v16bf r;
#pragma unroll
  for (int i = 0; i < 8; ++i) { r[i] = lo[i]; r[i + 8] = hi[i]; }
  return r;
}

static __device__ __forceinline__ v8f wmma_bf16(v16bf a, v16bf b, v8f c) {
  return __builtin_amdgcn_wmma_f32_16x16x32_bf16(false, a, false, b, (short)0, c,
                                                 false, false);
}

// ---------------------------------------------------------------------------
// fp32 -> bf16 convert (vectorized x4)
// ---------------------------------------------------------------------------
__global__ __launch_bounds__(256) void cvt_bf16(const float* __restrict__ in,
                                                __bf16* __restrict__ out,
                                                int n4) {
  int i = blockIdx.x * 256 + threadIdx.x;
  if (i < n4) {
    v4f f = *(const v4f*)(in + (size_t)i * 4);
#pragma unroll
    for (int j = 0; j < 4; ++j) out[(size_t)i * 4 + j] = (__bf16)f[j];
  }
}

// ---------------------------------------------------------------------------
// Tiled GEMM:  C[M,N] = A[M,K] @ W[N,K]^T + bias
// 128x128 block tile, K-step 32, 8 waves, double-buffered LDS, TDM staging.
// MODE 0: scatter bf16 into q/k/v [B,H,S,HD]   MODE 1: fp32 out [M,N]
// ---------------------------------------------------------------------------
#define LDT 40  // LDS row stride (elements); TDM pad: 16DW interval, 4DW pad

template <int MODE>
__global__ __launch_bounds__(256) void gemm_bf16_wmma(
    const __bf16* __restrict__ A, const __bf16* __restrict__ W,
    const float* __restrict__ bias, __bf16* __restrict__ oq,
    __bf16* __restrict__ ok, __bf16* __restrict__ ov, float* __restrict__ of32,
    int M, int N, int K) {
  __shared__ __bf16 As[2 * 128 * LDT];
  __shared__ __bf16 Bs[2 * 128 * LDT];
  const int BUFT = 128 * LDT;  // elements per buffer

  const int tid = threadIdx.x;
  const int wid = tid >> 5, lane = tid & 31, half = lane >> 4, lc = lane & 15;
  const int bm = blockIdx.y * 128, bn = blockIdx.x * 128;
  const int rowb = (wid & 3) * 32, colb = (wid >> 2) * 64;

  auto stage = [&](int buf, int k0) {
#if USE_TDM
    if (tid < 32) {
      tdm_load_2d(&A[(size_t)bm * K + k0], lds_off(&As[buf * BUFT]),
                  (unsigned)K, 128u, (unsigned)K, 32u, 128u, 3u, 3u);
      tdm_load_2d(&W[(size_t)bn * K + k0], lds_off(&Bs[buf * BUFT]),
                  (unsigned)K, 128u, (unsigned)K, 32u, 128u, 3u, 3u);
    }
#else
#pragma unroll
    for (int c = tid; c < 512; c += 256) {
      int r = c >> 2, cc = (c & 3) * 8;
      cp16(&As[buf * BUFT + r * LDT + cc],
           &A[(size_t)(bm + r) * K + k0 + cc]);
      cp16(&Bs[buf * BUFT + r * LDT + cc],
           &W[(size_t)(bn + r) * K + k0 + cc]);
    }
#endif
  };

  v8f acc[2][4];
#pragma unroll
  for (int mt = 0; mt < 2; ++mt)
#pragma unroll
    for (int nt = 0; nt < 4; ++nt) acc[mt][nt] = (v8f)(0.0f);

  stage(0, 0);
  stage_wait();
  __syncthreads();

  const int nk = K >> 5;
  for (int ki = 0; ki < nk; ++ki) {
    const int co = (ki & 1) * BUFT;  // current buffer element offset
    if (ki + 1 < nk) stage((ki & 1) ^ 1, (ki + 1) << 5);  // prefetch next tile

    v16bf af[2], bf[4];
#pragma unroll
    for (int mt = 0; mt < 2; ++mt) {
      int r = rowb + mt * 16 + lc;
      v8bf lo = *(v8bf*)&As[co + r * LDT + half * 8];
      v8bf hi = *(v8bf*)&As[co + r * LDT + 16 + half * 8];
      af[mt] = mk16(lo, hi);
    }
#pragma unroll
    for (int nt = 0; nt < 4; ++nt) {
      int n = colb + nt * 16 + lc;
      v8bf lo = *(v8bf*)&Bs[co + n * LDT + half * 8];
      v8bf hi = *(v8bf*)&Bs[co + n * LDT + 16 + half * 8];
      bf[nt] = mk16(lo, hi);
    }
#pragma unroll
    for (int mt = 0; mt < 2; ++mt)
#pragma unroll
      for (int nt = 0; nt < 4; ++nt)
        acc[mt][nt] = wmma_bf16(af[mt], bf[nt], acc[mt][nt]);

    stage_wait();
    __syncthreads();
  }

  // epilogue
#pragma unroll
  for (int mt = 0; mt < 2; ++mt) {
#pragma unroll
    for (int nt = 0; nt < 4; ++nt) {
#pragma unroll
      for (int g = 0; g < 8; ++g) {
        int m = bm + rowb + mt * 16 + g + half * 8;
        int e = bn + colb + nt * 16 + lc;
        float val = acc[mt][nt][g] + bias[e];
        if (MODE == 0) {
          int h = e / (3 * HDQ);
          int c = e - h * (3 * HDQ);
          int sel = c >> 6;
          int hd = c & 63;
          int b = m >> 11, s = m & 2047;
          __bf16* dst = (sel == 0) ? oq : ((sel == 1) ? ok : ov);
          dst[(((size_t)(b * HQ + h)) * SQ + s) * HDQ + hd] = (__bf16)val;
        } else {
          of32[(size_t)m * N + e] = val;
        }
      }
    }
  }
}

// ---------------------------------------------------------------------------
// Flash attention, one (b,h) per blockIdx.y, 128 query rows per block.
// Double-buffered K (TDM) and V (register-staged transpose).
// All LDS indexing is rooted in a single __shared__ base so every access
// lowers to ds_load/ds_store (no flat path).
// ---------------------------------------------------------------------------
#define LDK 72    // K-tile stride; TDM pad: 32DW interval (code 4), 4DW pad
#define LDVT 136  // V^T stride
#define LDSC 132  // score stride (floats)
#define LDP 136   // P stride

#define KS_ELEMS (128 * LDK)
#define VT_ELEMS (64 * LDVT)

#define ATTN_SMEM_BYTES                                                  \
  (2 * KS_ELEMS * 2 + 2 * VT_ELEMS * 2 + 128 * LDSC * 4 + 128 * LDP * 2 + \
   256 * 4)

__global__ __launch_bounds__(256) void attn_flash_wmma(
    const __bf16* __restrict__ q, const __bf16* __restrict__ k,
    const __bf16* __restrict__ v, __bf16* __restrict__ vals) {
  extern __shared__ char smem[];
  __bf16* Ks = (__bf16*)smem;                 // [2][128][LDK]
  __bf16* Vt = Ks + 2 * KS_ELEMS;             // [2][64][LDVT]
  float* Sc = (float*)(Vt + 2 * VT_ELEMS);    // [128][LDSC]
  __bf16* Ps = (__bf16*)(Sc + 128 * LDSC);    // [128][LDP]
  float* alpha_s = (float*)(Ps + 128 * LDP);  // [128]
  float* linv_s = alpha_s + 128;              // [128]

  const int tid = threadIdx.x;
  const int wid = tid >> 5, lane = tid & 31, half = lane >> 4, lc = lane & 15;
  const int bh = blockIdx.y;
  const int qbase = blockIdx.x * 128;
  const __bf16* qh = q + (size_t)bh * SQ * HDQ;
  const __bf16* kh = k + (size_t)bh * SQ * HDQ;
  const __bf16* vh = v + (size_t)bh * SQ * HDQ;

  auto stageK = [&](int buf, int kvb) {
#if USE_TDM
    if (tid < 32)
      tdm_load_2d(&kh[(size_t)kvb * HDQ], lds_off(&Ks[buf * KS_ELEMS]), HDQ,
                  128u, HDQ, HDQ, 128u, 4u, 3u);
#else
#pragma unroll
    for (int i = 0; i < 4; ++i) {
      int c = tid + i * 256;
      int r = c >> 3, cc = (c & 7) * 8;
      cp16(&Ks[buf * KS_ELEMS + r * LDK + cc],
           &kh[(size_t)(kvb + r) * HDQ + cc]);
    }
#endif
  };
  auto loadV = [&](v8bf* vv, int kvb) {
#pragma unroll
    for (int i = 0; i < 4; ++i) {
      int c = tid + i * 256;
      int r = c >> 3, cc = (c & 7) * 8;
      vv[i] = *(const v8bf*)&vh[(size_t)(kvb + r) * HDQ + cc];
    }
  };
  auto storeVt = [&](int buf, const v8bf* vv) {
#pragma unroll
    for (int i = 0; i < 4; ++i) {
      int c = tid + i * 256;
      int r = c >> 3, cc = (c & 7) * 8;
#pragma unroll
      for (int j = 0; j < 8; ++j)
        Vt[buf * VT_ELEMS + (cc + j) * LDVT + r] = vv[i][j];
    }
  };

  // Q fragments for this wave's 16 query rows (K=64 -> 2 chunks), from global.
  v16bf aq[2];
#pragma unroll
  for (int kc = 0; kc < 2; ++kc) {
    int r = qbase + wid * 16 + lc;
    v8bf lo = *(const v8bf*)&qh[(size_t)r * HDQ + kc * 32 + half * 8];
    v8bf hi = *(const v8bf*)&qh[(size_t)r * HDQ + kc * 32 + 16 + half * 8];
    aq[kc] = mk16(lo, hi);
  }

  v8f of[4];
#pragma unroll
  for (int nt = 0; nt < 4; ++nt) of[nt] = (v8f)(0.0f);
  float m_i = -3.0e38f;  // per-row stats live in threads 0..127
  float l_i = 0.0f;

  // prologue: stage tile 0
  {
    v8bf vv[4];
    stageK(0, 0);
    loadV(vv, 0);
    storeVt(0, vv);
    stage_wait();
    __syncthreads();
  }

  const int T = SQ / 128;
  for (int t = 0; t < T; ++t) {
    const int cur = t & 1;
    const int ko = cur * KS_ELEMS;  // current K buffer element offset
    const int vo = cur * VT_ELEMS;  // current V^T buffer element offset
    const bool nxt = (t + 1 < T);
    v8bf vv[4];
    if (nxt) {
      stageK(cur ^ 1, (t + 1) * 128);  // TDM/async in flight during compute
      loadV(vv, (t + 1) * 128);        // global loads overlap score WMMAs
    }

    // scores: Q[16x64] @ K_tile^T[64x128] -> 8 accum tiles per wave
    v8f sa[8];
#pragma unroll
    for (int nt = 0; nt < 8; ++nt) sa[nt] = (v8f)(0.0f);
#pragma unroll
    for (int kc = 0; kc < 2; ++kc) {
#pragma unroll
      for (int nt = 0; nt < 8; ++nt) {
        int n = nt * 16 + lc;
        v8bf lo = *(v8bf*)&Ks[ko + n * LDK + kc * 32 + half * 8];
        v8bf hi = *(v8bf*)&Ks[ko + n * LDK + kc * 32 + 16 + half * 8];
        sa[nt] = wmma_bf16(aq[kc], mk16(lo, hi), sa[nt]);
      }
    }
#pragma unroll
    for (int nt = 0; nt < 8; ++nt)
#pragma unroll
      for (int g = 0; g < 8; ++g)
        Sc[(wid * 16 + g + half * 8) * LDSC + nt * 16 + lc] =
            sa[nt][g] * 0.125f;  // 1/sqrt(64)
    if (nxt) storeVt(cur ^ 1, vv);
    __syncthreads();

    // online softmax: one thread per query row
    if (tid < 128) {
      float tmax = -3.0e38f;
      for (int j = 0; j < 128; ++j) tmax = fmaxf(tmax, Sc[tid * LDSC + j]);
      float mnew = fmaxf(m_i, tmax);
      float alpha = __expf(m_i - mnew);
      float lsum = 0.0f;
      for (int j = 0; j < 128; ++j) {
        float p = __expf(Sc[tid * LDSC + j] - mnew);
        lsum += p;
        Ps[tid * LDP + j] = (__bf16)p;
      }
      l_i = l_i * alpha + lsum;
      m_i = mnew;
      alpha_s[tid] = alpha;
    }
    __syncthreads();

    // rescale running O, then accumulate P @ V
#pragma unroll
    for (int g = 0; g < 8; ++g) {
      float a = alpha_s[wid * 16 + g + half * 8];
#pragma unroll
      for (int nt = 0; nt < 4; ++nt) of[nt][g] *= a;
    }
#pragma unroll
    for (int kc = 0; kc < 4; ++kc) {
      int r = wid * 16 + lc;
      v8bf lo = *(v8bf*)&Ps[r * LDP + kc * 32 + half * 8];
      v8bf hi = *(v8bf*)&Ps[r * LDP + kc * 32 + 16 + half * 8];
      v16bf ap = mk16(lo, hi);
#pragma unroll
      for (int nt = 0; nt < 4; ++nt) {
        int n = nt * 16 + lc;
        v8bf blo = *(v8bf*)&Vt[vo + n * LDVT + kc * 32 + half * 8];
        v8bf bhi = *(v8bf*)&Vt[vo + n * LDVT + kc * 32 + 16 + half * 8];
        of[nt] = wmma_bf16(ap, mk16(blo, bhi), of[nt]);
      }
    }
    stage_wait();
    __syncthreads();
  }

  if (tid < 128) linv_s[tid] = 1.0f / l_i;
  __syncthreads();

  const int b = bh >> 4, h = bh & 15;
#pragma unroll
  for (int nt = 0; nt < 4; ++nt) {
#pragma unroll
    for (int g = 0; g < 8; ++g) {
      int rl = wid * 16 + g + half * 8;
      int qr = qbase + rl;
      float val = of[nt][g] * linv_s[rl];
      vals[((size_t)(b * SQ + qr)) * DQ + h * HDQ + nt * 16 + lc] = (__bf16)val;
    }
  }
}

// ---------------------------------------------------------------------------
// Host launcher
// ---------------------------------------------------------------------------
extern "C" void kernel_launch(void* const* d_in, const int* in_sizes, int n_in,
                              void* d_out, int out_size, void* d_ws,
                              size_t ws_size, hipStream_t stream) {
  const float* x = (const float*)d_in[0];      // [B,S,D]
  const float* qkv_w = (const float*)d_in[1];  // [3D,D]
  const float* qkv_b = (const float*)d_in[2];  // [3D]
  const float* o_w = (const float*)d_in[3];    // [D,D]
  const float* o_b = (const float*)d_in[4];    // [D]
  float* out = (float*)d_out;                  // [B,S,D] fp32

  const int M = BQ * SQ;  // 4096
  size_t off = 0;
  auto carve = [&](size_t bytes) {
    void* p = (char*)d_ws + off;
    off += (bytes + 255) & ~(size_t)255;
    return p;
  };
  __bf16* xb = (__bf16*)carve((size_t)M * DQ * 2);
  __bf16* wqkv = (__bf16*)carve((size_t)3 * DQ * DQ * 2);
  __bf16* wo = (__bf16*)carve((size_t)DQ * DQ * 2);
  __bf16* qb = (__bf16*)carve((size_t)M * DQ * 2);
  __bf16* kb = (__bf16*)carve((size_t)M * DQ * 2);
  __bf16* vb = (__bf16*)carve((size_t)M * DQ * 2);
  __bf16* valb = (__bf16*)carve((size_t)M * DQ * 2);
  (void)ws_size; (void)n_in; (void)in_sizes; (void)out_size;

  {
    int n4 = (M * DQ) / 4;
    cvt_bf16<<<(n4 + 255) / 256, 256, 0, stream>>>(x, xb, n4);
    n4 = (3 * DQ * DQ) / 4;
    cvt_bf16<<<(n4 + 255) / 256, 256, 0, stream>>>(qkv_w, wqkv, n4);
    n4 = (DQ * DQ) / 4;
    cvt_bf16<<<(n4 + 255) / 256, 256, 0, stream>>>(o_w, wo, n4);
  }

  // QKV projection + head scatter
  gemm_bf16_wmma<0><<<dim3(3 * DQ / 128, M / 128), 256, 0, stream>>>(
      xb, wqkv, qkv_b, qb, kb, vb, nullptr, M, 3 * DQ, DQ);

  // flash attention
  static_assert(ATTN_SMEM_BYTES < 320 * 1024, "LDS budget");
  (void)hipFuncSetAttribute((const void*)attn_flash_wmma,
                            hipFuncAttributeMaxDynamicSharedMemorySize,
                            ATTN_SMEM_BYTES);
  attn_flash_wmma<<<dim3(SQ / 128, BQ * HQ), 256, ATTN_SMEM_BYTES, stream>>>(
      qb, kb, vb, valb);

  // output projection
  gemm_bf16_wmma<1><<<dim3(DQ / 128, M / 128), 256, 0, stream>>>(
      valb, wo, o_b, nullptr, nullptr, nullptr, out, M, DQ, DQ);
}
